// SparseSelfAttention_62955630624780
// MI455X (gfx1250) — matrix operand order repaired and checked by reading, not compiled, not run on hardware
//
#include <hip/hip_runtime.h>

// ---------------------------------------------------------------------------
// Sparse MoE self-attention for MI455X (gfx1250), bf16 WMMA pipeline.
// B=128, S=197, D=1024, HD=128, E=8, K=4.
// All GEMMs use v_wmma_f32_16x16x32_bf16 (fp32 accumulation).
// ---------------------------------------------------------------------------

#define BB   128
#define SS   197
#define DD   1024
#define HDH  128
#define EE   8
#define KK   4
#define BKN  (BB * KK)          // 512 routed (batch, expert) pairs
#define SPR  208                // S padded to 13*16 rows
#define SPC  224                // S padded to 7*32 for K-dim of P@V
#define MT   13                 // row tiles (208/16)
#define SD   (SS * DD)          // 201728, router reduction length

typedef __attribute__((ext_vector_type(8)))  __bf16 v8bf;
typedef __attribute__((ext_vector_type(16))) __bf16 v16bf;
typedef __attribute__((ext_vector_type(8)))  float  v8f;

static __device__ inline v16bf cat16(v8bf a, v8bf b) {
  return __builtin_shufflevector(a, b, 0,1,2,3,4,5,6,7,8,9,10,11,12,13,14,15);
}

// A-operand fragment (16x32 bf16, row-major source):
// lane l<16 : row = l,    elements = A[row][k0+0..7]  , A[row][k0+16..23]
// lane l>=16: row = l-16, elements = A[row][k0+8..15] , A[row][k0+24..31]
static __device__ inline v16bf load_a_frag(const __bf16* row, int k0, bool hi) {
  int o = k0 + (hi ? 8 : 0);
  v8bf c0 = *(const v8bf*)(row + o);
  v8bf c1 = *(const v8bf*)(row + o + 16);
  return cat16(c0, c1);
}

// B-operand fragment (32x16 bf16) from a column-contiguous layout:
// lane l<16 : col = l,    elements = Bt[col][k0+0..7]  , Bt[col][k0+16..23]
// lane l>=16: col = l-16, elements = Bt[col][k0+8..15] , Bt[col][k0+24..31]
static __device__ inline v16bf load_b_frag(const __bf16* col, int k0, bool hi) {
  int o = k0 + (hi ? 8 : 0);
  v8bf c0 = *(const v8bf*)(col + o);
  v8bf c1 = *(const v8bf*)(col + o + 16);
  return cat16(c0, c1);
}

static __device__ inline v8f wmma_bf16(v16bf a, v16bf b, v8f c) {
  return __builtin_amdgcn_wmma_f32_16x16x32_bf16(false, a, false, b,
                                                 (short)0, c, false, false);
}

// ---------------------------------------------------------------------------
// K1: router. logits[b,e] = X[b].flat . W_switch[:,e] + b_switch[e]; top-4 ids.
// ---------------------------------------------------------------------------
__global__ __launch_bounds__(256)
void router_kernel(const float* __restrict__ X, const float* __restrict__ Wsw,
                   const float* __restrict__ bsw, int* __restrict__ ids) {
  __shared__ float red[256][EE];
  int b = blockIdx.x, t = threadIdx.x;
  float acc[EE];
#pragma unroll
  for (int e = 0; e < EE; ++e) acc[e] = 0.0f;
  const float* xb = X + (size_t)b * SD;
  for (int i = t; i < SD; i += 256) {
    float x = xb[i];
    const float* w = Wsw + (size_t)i * EE;
#pragma unroll
    for (int e = 0; e < EE; ++e) acc[e] += x * w[e];
  }
#pragma unroll
  for (int e = 0; e < EE; ++e) red[t][e] = acc[e];
  __syncthreads();
  for (int st = 128; st > 0; st >>= 1) {
    if (t < st) {
#pragma unroll
      for (int e = 0; e < EE; ++e) red[t][e] += red[t + st][e];
    }
    __syncthreads();
  }
  if (t == 0) {
    float lg[EE];
    bool used[EE];
#pragma unroll
    for (int e = 0; e < EE; ++e) { lg[e] = red[0][e] + bsw[e]; used[e] = false; }
    // top-k of logits == top-k of softmax (monotonic); ties -> lowest index
    for (int k = 0; k < KK; ++k) {
      int best = 0; float bv = -3.4e38f;
      for (int e = 0; e < EE; ++e)
        if (!used[e] && lg[e] > bv) { bv = lg[e]; best = e; }
      used[best] = true;
      ids[b * KK + k] = best;
    }
  }
}

// ---------------------------------------------------------------------------
// K2: X fp32 -> bf16
// ---------------------------------------------------------------------------
__global__ __launch_bounds__(256)
void cvt_x_kernel(const float* __restrict__ X, __bf16* __restrict__ Xh) {
  const long long n = (long long)BB * SD;
  for (long long i = (long long)blockIdx.x * blockDim.x + threadIdx.x; i < n;
       i += (long long)gridDim.x * blockDim.x)
    Xh[i] = (__bf16)X[i];
}

// ---------------------------------------------------------------------------
// K3: weights -> bf16 with transpose.
// Wqkvt[mat][e][h][d] = W{q,k,v}[e][d][h] ; Wft[e][d][h] = Wf[e][h][d]
// ---------------------------------------------------------------------------
__global__ __launch_bounds__(256)
void cvt_w_kernel(const float* __restrict__ Wq, const float* __restrict__ Wk,
                  const float* __restrict__ Wv, const float* __restrict__ Wf,
                  __bf16* __restrict__ Wqkvt, __bf16* __restrict__ Wft) {
  const int per = EE * DD * HDH;           // 1,048,576
  const long long n = 4LL * per;
  for (long long i = (long long)blockIdx.x * blockDim.x + threadIdx.x; i < n;
       i += (long long)gridDim.x * blockDim.x) {
    int mat = (int)(i / per);
    int r   = (int)(i % per);
    int e   = r / (DD * HDH);
    int rr  = r % (DD * HDH);
    if (mat < 3) {                          // dst [e][h][d] <- src [e][d][h]
      int h = rr / DD, d = rr % DD;
      const float* src = (mat == 0) ? Wq : (mat == 1) ? Wk : Wv;
      Wqkvt[i] = (__bf16)src[((size_t)e * DD + d) * HDH + h];
    } else {                                // dst [e][d][h] <- src [e][h][d]
      int d = rr / HDH, h = rr % HDH;
      Wft[r] = (__bf16)Wf[((size_t)e * HDH + h) * DD + d];
    }
  }
}

// ---------------------------------------------------------------------------
// K4: zero Vt tail columns [208,224) (cols 197..207 get finite clamped values
// from K5; P columns >=197 are exactly zero so those products vanish).
// ---------------------------------------------------------------------------
__global__ __launch_bounds__(256)
void zero_vt_tail_kernel(__bf16* __restrict__ Vt) {
  const long long n = (long long)BKN * HDH * 16;
  for (long long i = (long long)blockIdx.x * blockDim.x + threadIdx.x; i < n;
       i += (long long)gridDim.x * blockDim.x) {
    int j  = (int)(i & 15);
    long long r = i >> 4;                   // bk*HDH + h
    Vt[r * SPC + 208 + j] = (__bf16)0.0f;
  }
}

// ---------------------------------------------------------------------------
// K5: QKV projection. One wave per (bk, row-tile, h-tile). Row index is
// CLAMPED (no divergent exec masking); padded rows carry duplicate values
// that only reach provably-dead regions. Ping-pong double buffering: each
// fragment set is loaded in place and consumed directly by WMMA (no register
// rotation copies); last iteration peeled so the steady-state loop is
// branch-free.
// Q, Km row-major [SPR][HDH]; V stored transposed Vt [HDH][SPC].
// ---------------------------------------------------------------------------
__global__ __launch_bounds__(32)
void qkv_kernel(const __bf16* __restrict__ Xh, const __bf16* __restrict__ Wqkvt,
                const int* __restrict__ ids, __bf16* __restrict__ Q,
                __bf16* __restrict__ Km, __bf16* __restrict__ Vt) {
  int bk = blockIdx.x, mt = blockIdx.y, ht = blockIdx.z;
  int b = bk >> 2;
  int e = ids[bk];
  int l = threadIdx.x;
  bool hi = l >= 16;
  int lm = l & 15;

  int srow = mt * 16 + lm;
  int crow = srow < SS ? srow : SS - 1;     // clamp: no divergence
  const __bf16* xrow = Xh + (size_t)b * SD + (size_t)crow * DD;
  int hcol = ht * 16 + lm;
  const __bf16* wq = Wqkvt + ((size_t)(0 * EE + e) * HDH + hcol) * DD;
  const __bf16* wk = Wqkvt + ((size_t)(1 * EE + e) * HDH + hcol) * DD;
  const __bf16* wv = Wqkvt + ((size_t)(2 * EE + e) * HDH + hcol) * DD;

  v8f cq = {}, ck = {}, cv = {};
  // set 0 <- chunk 0
  v16bf a0  = load_a_frag(xrow, 0, hi);
  v16bf q0  = load_b_frag(wq, 0, hi);
  v16bf k0f = load_b_frag(wk, 0, hi);
  v16bf v0f = load_b_frag(wv, 0, hi);
#pragma unroll 1
  for (int kc = 0; kc < DD / 32 - 2; kc += 2) {
    int kb1 = (kc + 1) * 32;
    v16bf a1  = load_a_frag(xrow, kb1, hi);     // set 1 <- chunk kc+1
    v16bf q1  = load_b_frag(wq, kb1, hi);
    v16bf k1f = load_b_frag(wk, kb1, hi);
    v16bf v1f = load_b_frag(wv, kb1, hi);
    cq = wmma_bf16(a0, q0, cq);                 // consume set 0 (chunk kc)
    ck = wmma_bf16(a0, k0f, ck);
    cv = wmma_bf16(a0, v0f, cv);
    int kb2 = (kc + 2) * 32;
    a0  = load_a_frag(xrow, kb2, hi);           // set 0 <- chunk kc+2
    q0  = load_b_frag(wq, kb2, hi);
    k0f = load_b_frag(wk, kb2, hi);
    v0f = load_b_frag(wv, kb2, hi);
    cq = wmma_bf16(a1, q1, cq);                 // consume set 1 (chunk kc+1)
    ck = wmma_bf16(a1, k1f, ck);
    cv = wmma_bf16(a1, v1f, cv);
  }
  {                                             // epilogue: chunks N-2, N-1
    int kb1 = (DD / 32 - 1) * 32;
    v16bf a1  = load_a_frag(xrow, kb1, hi);
    v16bf q1  = load_b_frag(wq, kb1, hi);
    v16bf k1f = load_b_frag(wk, kb1, hi);
    v16bf v1f = load_b_frag(wv, kb1, hi);
    cq = wmma_bf16(a0, q0, cq);
    ck = wmma_bf16(a0, k0f, ck);
    cv = wmma_bf16(a0, v0f, cv);
    cq = wmma_bf16(a1, q1, cq);
    ck = wmma_bf16(a1, k1f, ck);
    cv = wmma_bf16(a1, v1f, cv);
  }

  __bf16* qb = Q  + (size_t)bk * SPR * HDH;
  __bf16* kb = Km + (size_t)bk * SPR * HDH;
  __bf16* vb = Vt + (size_t)bk * HDH * SPC;
#pragma unroll
  for (int g = 0; g < 8; ++g) {
    int r = mt * 16 + g + (hi ? 8 : 0);
    qb[(size_t)r * HDH + hcol] = (__bf16)cq[g];
    kb[(size_t)r * HDH + hcol] = (__bf16)ck[g];
    vb[(size_t)hcol * SPC + r] = (__bf16)cv[g];
  }
}

// ---------------------------------------------------------------------------
// K6: attention core. One wave per (bk, 16-row block):
//   scores(16x208) via WMMA -> LDS fp32, masked softmax (197 cols, all 32
//   lanes: 2 lanes per row combined via shfl_xor), P bf16 in LDS
//   (cols 197..223 zero), O = P@V via WMMA -> O [SPR][HDH].
// ---------------------------------------------------------------------------
__global__ __launch_bounds__(32)
void attn_kernel(const __bf16* __restrict__ Q, const __bf16* __restrict__ Km,
                 const __bf16* __restrict__ Vt, __bf16* __restrict__ O) {
  __shared__ float  sc[16][228];   // stride padded vs 64-bank LDS
  __shared__ __bf16 pb[16][232];   // stride mult-of-8 keeps 16B-aligned chunks
  int bk = blockIdx.x, mt = blockIdx.y;
  int l = threadIdx.x;
  bool hi = l >= 16;
  int lm = l & 15;
  const float scale = 0.08838834764831845f;   // 1/sqrt(128)

  const __bf16* qrow = Q + (size_t)bk * SPR * HDH + (size_t)(mt * 16 + lm) * HDH;
  v16bf aq[4];
#pragma unroll
  for (int c = 0; c < 4; ++c) aq[c] = load_a_frag(qrow, c * 32, hi);

  const __bf16* kb = Km + (size_t)bk * SPR * HDH;
#pragma unroll 1
  for (int nt = 0; nt < MT; ++nt) {
    const __bf16* kcol = kb + (size_t)(nt * 16 + lm) * HDH;
    v16bf bfr[4];
#pragma unroll
    for (int kc = 0; kc < 4; ++kc) bfr[kc] = load_b_frag(kcol, kc * 32, hi);
    v8f c = {};
#pragma unroll
    for (int kc = 0; kc < 4; ++kc) c = wmma_bf16(aq[kc], bfr[kc], c);
#pragma unroll
    for (int g = 0; g < 8; ++g)
      sc[g + (hi ? 8 : 0)][nt * 16 + lm] = c[g] * scale;
  }
  __syncthreads();

  {
    // two lanes per row: lane lm covers [0,112), lane lm+16 covers [112,197)
    int c0 = hi ? 112 : 0;
    int c1 = hi ? SS : 112;
    float m = -3.4e38f;
    for (int t = c0; t < c1; ++t) m = fmaxf(m, sc[lm][t]);
    m = fmaxf(m, __shfl_xor(m, 16));
    float s = 0.0f;
    for (int t = c0; t < c1; ++t) {
      float ev = __expf(sc[lm][t] - m);
      sc[lm][t] = ev;
      s += ev;
    }
    s += __shfl_xor(s, 16);
    float inv = 1.0f / s;
    for (int t = c0; t < c1; ++t) pb[lm][t] = (__bf16)(sc[lm][t] * inv);
    if (hi)
      for (int t = SS; t < SPC; ++t) pb[lm][t] = (__bf16)0.0f;
  }
  __syncthreads();

  const __bf16* vtb = Vt + (size_t)bk * HDH * SPC;
  __bf16* ob = O + (size_t)bk * SPR * HDH;
  const __bf16* prow = &pb[lm][0];
#pragma unroll 1
  for (int ht = 0; ht < HDH / 16; ++ht) {
    const __bf16* vcol = vtb + (size_t)(ht * 16 + lm) * SPC;
    v16bf bfr[7];
#pragma unroll
    for (int kc = 0; kc < SPC / 32; ++kc) bfr[kc] = load_b_frag(vcol, kc * 32, hi);
    v8f c = {};
#pragma unroll
    for (int kc = 0; kc < SPC / 32; ++kc)
      c = wmma_bf16(load_a_frag(prow, kc * 32, hi), bfr[kc], c);
#pragma unroll
    for (int g = 0; g < 8; ++g)
      ob[(size_t)(mt * 16 + g + (hi ? 8 : 0)) * HDH + ht * 16 + lm] = (__bf16)c[g];
  }
}

// ---------------------------------------------------------------------------
// K7: output projection + expert-sum. One wave per 16x32 out tile (2 C tiles,
// A fragments reused across both); loops the 4 routed experts; C seeded with
// 4*b_f (reference adds b_f once per expert). Fragments batch-loaded so loads
// clause ahead of the WMMA burst.
// ---------------------------------------------------------------------------
__global__ __launch_bounds__(32)
void out_kernel(const __bf16* __restrict__ O, const __bf16* __restrict__ Wft,
                const int* __restrict__ ids, const float* __restrict__ bf,
                float* __restrict__ out) {
  int b = blockIdx.x, mt = blockIdx.y, nt = blockIdx.z;
  int l = threadIdx.x;
  bool hi = l >= 16;
  int lm = l & 15;
  int d0 = nt * 32 + lm;
  int d1 = d0 + 16;

  float bi0 = 4.0f * bf[d0];
  float bi1 = 4.0f * bf[d1];
  v8f c0, c1;
#pragma unroll
  for (int g = 0; g < 8; ++g) { c0[g] = bi0; c1[g] = bi1; }

#pragma unroll 1
  for (int k = 0; k < KK; ++k) {
    int bk = b * KK + k;
    int e = ids[bk];
    const __bf16* orow  = O + (size_t)bk * SPR * HDH + (size_t)(mt * 16 + lm) * HDH;
    const __bf16* wcol0 = Wft + ((size_t)e * DD + d0) * HDH;
    const __bf16* wcol1 = Wft + ((size_t)e * DD + d1) * HDH;
    v16bf afr[4], b0f[4], b1f[4];
#pragma unroll
    for (int kc = 0; kc < 4; ++kc) {
      afr[kc] = load_a_frag(orow, kc * 32, hi);
      b0f[kc] = load_b_frag(wcol0, kc * 32, hi);
      b1f[kc] = load_b_frag(wcol1, kc * 32, hi);
    }
#pragma unroll
    for (int kc = 0; kc < 4; ++kc) {
      c0 = wmma_bf16(afr[kc], b0f[kc], c0);
      c1 = wmma_bf16(afr[kc], b1f[kc], c1);
    }
  }
#pragma unroll
  for (int g = 0; g < 8; ++g) {
    int s = mt * 16 + g + (hi ? 8 : 0);
    if (s < SS) {
      out[((size_t)b * SS + s) * DD + d0] = c0[g];
      out[((size_t)b * SS + s) * DD + d1] = c1[g];
    }
  }
}

// ---------------------------------------------------------------------------
extern "C" void kernel_launch(void* const* d_in, const int* in_sizes, int n_in,
                              void* d_out, int out_size, void* d_ws, size_t ws_size,
                              hipStream_t stream) {
  const float* X   = (const float*)d_in[0];
  const float* Wsw = (const float*)d_in[1];
  const float* bsw = (const float*)d_in[2];
  const float* Wq  = (const float*)d_in[3];
  const float* Wk  = (const float*)d_in[4];
  const float* Wv  = (const float*)d_in[5];
  const float* Wf  = (const float*)d_in[6];
  const float* bf  = (const float*)d_in[7];
  float* out = (float*)d_out;

  // Workspace carve (aligned to 256 B). Total ~171 MB.
  size_t off = 0;
  auto carve = [&](size_t bytes) -> char* {
    char* p = (char*)d_ws + off;
    off += (bytes + 255) & ~(size_t)255;
    return p;
  };
  __bf16* Xh    = (__bf16*)carve((size_t)BB * SD * 2);
  __bf16* Wqkvt = (__bf16*)carve((size_t)3 * EE * DD * HDH * 2);
  __bf16* Wft   = (__bf16*)carve((size_t)EE * DD * HDH * 2);
  __bf16* Q     = (__bf16*)carve((size_t)BKN * SPR * HDH * 2);
  __bf16* Km    = (__bf16*)carve((size_t)BKN * SPR * HDH * 2);
  __bf16* Vt    = (__bf16*)carve((size_t)BKN * HDH * SPC * 2);
  __bf16* Obuf  = (__bf16*)carve((size_t)BKN * SPR * HDH * 2);
  int*    ids   = (int*)   carve((size_t)BKN * 4);
  (void)in_sizes; (void)n_in; (void)out_size; (void)ws_size;

  cvt_x_kernel<<<4096, 256, 0, stream>>>(X, Xh);
  cvt_w_kernel<<<2048, 256, 0, stream>>>(Wq, Wk, Wv, Wf, Wqkvt, Wft);
  router_kernel<<<BB, 256, 0, stream>>>(X, Wsw, bsw, ids);
  zero_vt_tail_kernel<<<1024, 256, 0, stream>>>(Vt);
  qkv_kernel<<<dim3(BKN, MT, HDH / 16), 32, 0, stream>>>(Xh, Wqkvt, ids, Q, Km, Vt);
  attn_kernel<<<dim3(BKN, MT), 32, 0, stream>>>(Q, Km, Vt, Obuf);
  out_kernel<<<dim3(BB, MT, DD / 32), 32, 0, stream>>>(Obuf, Wft, ids, bf, out);
}
